// MultiHeadAttention_7748121002013
// MI455X (gfx1250) — matrix operand order repaired and checked by reading, not compiled, not run on hardware
//
#include <hip/hip_runtime.h>
#include <hip/hip_bf16.h>

typedef __attribute__((ext_vector_type(16))) __bf16 v16bf;
typedef __attribute__((ext_vector_type(8)))  float  v8f;

namespace {
constexpr int B_  = 2;
constexpr int S_  = 2048;
constexpr int D_  = 1024;
constexpr int H_  = 16;
constexpr int DH_ = 64;
constexpr int M_  = B_ * S_;   // 4096
}

__device__ __forceinline__ unsigned short f2bf(float x) {
  unsigned int u = __float_as_uint(x);
  unsigned int r = (u + 0x7FFFu + ((u >> 16) & 1u)) >> 16;  // round-nearest-even
  return (unsigned short)r;
}

union Frag {
  v16bf v;
  uint4 q[2];
};

// ---------------------------------------------------------------------------
// fp32 -> bf16 streaming convert (same layout), float4 per thread.
// ---------------------------------------------------------------------------
__global__ __launch_bounds__(256) void convert_bf16(
    const float* __restrict__ x, unsigned short* __restrict__ y)
{
  const size_t i = ((size_t)blockIdx.x * 256 + threadIdx.x) * 4;
  const float4 f = *(const float4*)(x + i);
  y[i + 0] = f2bf(f.x); y[i + 1] = f2bf(f.y);
  y[i + 2] = f2bf(f.z); y[i + 3] = f2bf(f.w);
}

// ---------------------------------------------------------------------------
// W[K,N] fp32 -> Wt[N,K] bf16 (LDS tile transpose, 32x32).
// ---------------------------------------------------------------------------
__global__ __launch_bounds__(256) void transpose_to_bf16(
    const float* __restrict__ W, unsigned short* __restrict__ Wt, int K, int N)
{
  __shared__ unsigned short tile[32][33];
  const int k0 = blockIdx.y * 32, n0 = blockIdx.x * 32;
  const int tx = threadIdx.x & 31, ty = threadIdx.x >> 5;
  for (int r = ty; r < 32; r += 8)
    tile[r][tx] = f2bf(W[(size_t)(k0 + r) * N + n0 + tx]);
  __syncthreads();
  for (int r = ty; r < 32; r += 8)
    Wt[(size_t)(n0 + r) * K + k0 + tx] = tile[tx][r];
}

// ---------------------------------------------------------------------------
// bf16 GEMM, fragments loaded straight from global (L2-resident operands).
//   C[M,N] = A[M,K] @ Wt[N,K]^T + bias, stored per `mode`:
//   mode 0: bf16 -> [B,H,S,DH]   (Q, 1/sqrt(DH) folded in)
//   mode 1: bf16 -> [B,H,S,DH]   (K)
//   mode 2: bf16 -> [B,H,DH,S]   (V transposed for the P@V B-operand)
//   mode 3: fp32 -> [M,N]        (final output)
// 128 threads = 4 waves; block tile 128x64; wave tile 32x64 (8 WMMA / k-step).
// ---------------------------------------------------------------------------
__global__ __launch_bounds__(128) void gemm_bf16_direct(
    const unsigned short* __restrict__ Abf, const unsigned short* __restrict__ Wt,
    const float* __restrict__ bias, void* __restrict__ outp,
    int M, int K, int N, int mode, float scale)
{
  const int t    = threadIdx.x;
  const int lane = t & 31;
  const int wid  = t >> 5;
  const int nl   = lane & 15;
  const int hl   = lane >> 4;
  const int m0   = blockIdx.y * 128;
  const int n0   = blockIdx.x * 64;

  // A operand: lane = row, interleaved K chunks {8h..8h+7, 16+8h..23+8h}.
  const unsigned short* a0 = Abf + (size_t)(m0 + wid * 32 + nl) * K;
  const unsigned short* a1 = a0 + (size_t)16 * K;
  // B operand: lane = output column, interleaved K chunks {8h..8h+7, 16+8h..23+8h}.
  const unsigned short* bp[4];
#pragma unroll
  for (int ct = 0; ct < 4; ++ct)
    bp[ct] = Wt + (size_t)(n0 + ct * 16 + nl) * K + 8 * hl;

  v8f acc[2][4] = {};

  for (int k0 = 0; k0 < K; k0 += 32) {
    Frag a[2], b[4];
    a[0].q[0] = *(const uint4*)(a0 + k0 + 8 * hl);
    a[0].q[1] = *(const uint4*)(a0 + k0 + 16 + 8 * hl);
    a[1].q[0] = *(const uint4*)(a1 + k0 + 8 * hl);
    a[1].q[1] = *(const uint4*)(a1 + k0 + 16 + 8 * hl);
#pragma unroll
    for (int ct = 0; ct < 4; ++ct) {
      b[ct].q[0] = *(const uint4*)(bp[ct] + k0);
      b[ct].q[1] = *(const uint4*)(bp[ct] + k0 + 16);
    }
    if (k0 + 32 < K) {  // L2/WGP$ prefetch of next k-lines (global_prefetch_b8)
      __builtin_prefetch(a0 + k0 + 32, 0, 1);
      __builtin_prefetch(a1 + k0 + 32, 0, 1);
      __builtin_prefetch(bp[0] + k0 + 32, 0, 1);
    }
#pragma unroll
    for (int r = 0; r < 2; ++r)
#pragma unroll
      for (int ct = 0; ct < 4; ++ct)
        acc[r][ct] = __builtin_amdgcn_wmma_f32_16x16x32_bf16(
            false, a[r].v, false, b[ct].v, (short)0, acc[r][ct], false, false);
  }

#pragma unroll
  for (int r = 0; r < 2; ++r) {
#pragma unroll
    for (int ct = 0; ct < 4; ++ct) {
#pragma unroll
      for (int i = 0; i < 8; ++i) {
        const int gm = m0 + wid * 32 + r * 16 + i + 8 * hl;  // C row = i + 8*(lane/16)
        const int gn = n0 + ct * 16 + nl;                    // C col = lane%16
        const float val = (acc[r][ct][i] + bias[gn]) * scale;
        if (mode == 3) {
          ((float*)outp)[(size_t)gm * N + gn] = val;
        } else {
          const int bb = gm >> 11, s = gm & (S_ - 1);
          const int h  = gn >> 6,  d = gn & (DH_ - 1);
          unsigned short* o = (unsigned short*)outp;
          const size_t idx = (mode == 2)
              ? (((size_t)(bb * H_ + h) * DH_ + d) * S_ + s)   // V^T
              : (((size_t)(bb * H_ + h) * S_ + s) * DH_ + d);  // Q / K
          o[idx] = f2bf(val);
        }
      }
    }
  }
}

// ---------------------------------------------------------------------------
// Attention: one workgroup = 16 query rows of one (b,h).
// scores (128 KB fp32) + P (64 KB bf16) live in dynamic LDS.
// ---------------------------------------------------------------------------
__global__ __launch_bounds__(128) void attn_kernel(
    const unsigned short* __restrict__ Qb, const unsigned short* __restrict__ Kb,
    const unsigned short* __restrict__ Vt, const unsigned char* __restrict__ mask,
    float* __restrict__ attn, unsigned short* __restrict__ ctxb)
{
  extern __shared__ char smem[];
  float          (*sc)[S_] = (float (*)[S_])smem;                                  // 16*2048 f32
  unsigned short (*pb)[S_] = (unsigned short (*)[S_])(smem + (size_t)16 * S_ * 4); // 16*2048 bf16
  float* red   = (float*)(smem + (size_t)16 * S_ * 4 + (size_t)16 * S_ * 2);       // 128
  float* rstat = red + 128;                                                        // 32

  const int t    = threadIdx.x;
  const int lane = t & 31;
  const int wid  = t >> 5;
  const int nl   = lane & 15;
  const int hl   = lane >> 4;
  const int b  = blockIdx.z, h = blockIdx.y, q0 = blockIdx.x * 16;
  const size_t bh = (size_t)(b * H_ + h);

  // Q fragments for rows q0..q0+15 (scale already folded in), kept in VGPRs.
  Frag qf[2];
  {
    const unsigned short* qp = Qb + (bh * S_ + q0 + nl) * DH_;
#pragma unroll
    for (int kk = 0; kk < 2; ++kk) {
      qf[kk].q[0] = *(const uint4*)(qp + kk * 32 + 8 * hl);
      qf[kk].q[1] = *(const uint4*)(qp + kk * 32 + 16 + 8 * hl);
    }
  }

  // scores = Q @ K^T (already scaled), masked, into LDS. Waves split column tiles.
  for (int ct = wid; ct < S_ / 16; ct += 4) {
    const int c0 = ct * 16;
    const unsigned short* kp = Kb + (bh * S_ + c0 + nl) * DH_;  // lane = score col = K row
    Frag kf0, kf1;
    kf0.q[0] = *(const uint4*)(kp + 8 * hl);
    kf0.q[1] = *(const uint4*)(kp + 16 + 8 * hl);
    kf1.q[0] = *(const uint4*)(kp + 32 + 8 * hl);
    kf1.q[1] = *(const uint4*)(kp + 48 + 8 * hl);
    v8f acc = {};
    acc = __builtin_amdgcn_wmma_f32_16x16x32_bf16(false, qf[0].v, false, kf0.v, (short)0, acc, false, false);
    acc = __builtin_amdgcn_wmma_f32_16x16x32_bf16(false, qf[1].v, false, kf1.v, (short)0, acc, false, false);
#pragma unroll
    for (int i = 0; i < 8; ++i) {
      const int qr = i + 8 * hl;
      const int kc = c0 + nl;
      float v = acc[i];
      if (mask[((size_t)b * S_ + q0 + qr) * S_ + kc]) v = -1e9f;
      sc[qr][kc] = v;
    }
  }
  __syncthreads();

  // Softmax: 8 threads per row, 256 columns each.
  const int r = t >> 3, seg = t & 7;
  const int cbase = seg * 256;
  float mmax = -3.0e38f;
  for (int c = 0; c < 256; ++c) mmax = fmaxf(mmax, sc[r][cbase + c]);
  red[r * 8 + seg] = mmax;
  __syncthreads();
  if (seg == 0) {
    float mm = red[r * 8];
    for (int j = 1; j < 8; ++j) mm = fmaxf(mm, red[r * 8 + j]);
    rstat[r] = mm;
  }
  __syncthreads();
  const float mm = rstat[r];
  float ssum = 0.f;
  for (int c = 0; c < 256; ++c) ssum += __expf(sc[r][cbase + c] - mm);
  red[r * 8 + seg] = ssum;
  __syncthreads();
  if (seg == 0) {
    float s = red[r * 8];
    for (int j = 1; j < 8; ++j) s += red[r * 8 + j];
    rstat[16 + r] = 1.0f / s;
  }
  __syncthreads();
  const float inv = rstat[16 + r];
  float* arow = attn + (bh * S_ + q0 + r) * S_;
  for (int c = 0; c < 256; ++c) {
    const float p = __expf(sc[r][cbase + c] - mm) * inv;
    arow[cbase + c] = p;          // attn_prob output (fp32)
    pb[r][cbase + c] = f2bf(p);   // bf16 P for the P@V WMMA
  }
  __syncthreads();

  // context[16x64] = P[16x2048] @ V[2048x64]; wave `wid` owns d-tile wid*16..+15.
  v8f cacc = {};
  const unsigned short* vp = Vt + (bh * DH_ + wid * 16 + nl) * S_;  // V^T: contiguous K/lane
  for (int k0 = 0; k0 < S_; k0 += 32) {
    Frag pf, vf;
    pf.q[0] = *(const uint4*)&pb[nl][k0 + 8 * hl];
    pf.q[1] = *(const uint4*)&pb[nl][k0 + 16 + 8 * hl];
    vf.q[0] = *(const uint4*)(vp + k0 + 8 * hl);
    vf.q[1] = *(const uint4*)(vp + k0 + 16 + 8 * hl);
    cacc = __builtin_amdgcn_wmma_f32_16x16x32_bf16(false, pf.v, false, vf.v, (short)0, cacc, false, false);
  }
#pragma unroll
  for (int i = 0; i < 8; ++i) {
    const int qr = i + 8 * hl;
    const int d  = wid * 16 + nl;
    ctxb[((size_t)b * S_ + q0 + qr) * D_ + h * DH_ + d] = f2bf(cacc[i]);  // bf16 context
  }
}

// ---------------------------------------------------------------------------
extern "C" void kernel_launch(void* const* d_in, const int* in_sizes, int n_in,
                              void* d_out, int out_size, void* d_ws, size_t ws_size,
                              hipStream_t stream) {
  const float* XQ = (const float*)d_in[0];
  const float* XK = (const float*)d_in[1];
  const float* XV = (const float*)d_in[2];
  const unsigned char* mask = (const unsigned char*)d_in[3];
  const float* Wq = (const float*)d_in[4];  const float* bq = (const float*)d_in[5];
  const float* Wk = (const float*)d_in[6];  const float* bk = (const float*)d_in[7];
  const float* Wv = (const float*)d_in[8];  const float* bv = (const float*)d_in[9];
  const float* Wo = (const float*)d_in[10]; const float* bo = (const float*)d_in[11];

  char* ws = (char*)d_ws;
  const size_t MB = 1024ull * 1024ull;
  unsigned short* Xqb  = (unsigned short*)(ws + 0 * MB);   // [M,D] bf16, 8MB
  unsigned short* Xkb  = (unsigned short*)(ws + 8 * MB);
  unsigned short* Xvb  = (unsigned short*)(ws + 16 * MB);
  unsigned short* Wtq  = (unsigned short*)(ws + 24 * MB);  // [N,K] bf16, 2MB each
  unsigned short* Wtk  = (unsigned short*)(ws + 26 * MB);
  unsigned short* Wtv  = (unsigned short*)(ws + 28 * MB);
  unsigned short* Wto  = (unsigned short*)(ws + 30 * MB);
  unsigned short* Qb   = (unsigned short*)(ws + 32 * MB);  // [B,H,S,DH] bf16
  unsigned short* Kb   = (unsigned short*)(ws + 40 * MB);
  unsigned short* Vt   = (unsigned short*)(ws + 48 * MB);  // [B,H,DH,S] bf16
  unsigned short* ctxb = (unsigned short*)(ws + 56 * MB);  // [M,D] bf16

  float* out  = (float*)d_out;              // [B,S,D] fp32
  float* attn = out + (size_t)M_ * D_;      // [B,H,S,S] fp32

  const int cvt_blocks = (int)(((size_t)M_ * D_) / 1024);  // 4 floats/thread, 256 thr
  convert_bf16<<<cvt_blocks, 256, 0, stream>>>(XQ, Xqb);
  convert_bf16<<<cvt_blocks, 256, 0, stream>>>(XK, Xkb);
  convert_bf16<<<cvt_blocks, 256, 0, stream>>>(XV, Xvb);

  const dim3 gt(D_ / 32, D_ / 32, 1);
  transpose_to_bf16<<<gt, 256, 0, stream>>>(Wq, Wtq, D_, H_ * DH_);
  transpose_to_bf16<<<gt, 256, 0, stream>>>(Wk, Wtk, D_, H_ * DH_);
  transpose_to_bf16<<<gt, 256, 0, stream>>>(Wv, Wtv, D_, H_ * DH_);
  transpose_to_bf16<<<gt, 256, 0, stream>>>(Wo, Wto, H_ * DH_, D_);

  const dim3 gp((H_ * DH_) / 64, M_ / 128, 1);
  gemm_bf16_direct<<<gp, 128, 0, stream>>>(Xqb, Wtq, bq, Qb, M_, D_, H_ * DH_, 0, 0.125f);
  gemm_bf16_direct<<<gp, 128, 0, stream>>>(Xkb, Wtk, bk, Kb, M_, D_, H_ * DH_, 1, 1.0f);
  gemm_bf16_direct<<<gp, 128, 0, stream>>>(Xvb, Wtv, bv, Vt, M_, D_, H_ * DH_, 2, 1.0f);

  const size_t lds = (size_t)16 * S_ * 4 + (size_t)16 * S_ * 2 + (128 + 32) * sizeof(float);
  hipFuncSetAttribute((const void*)attn_kernel,
                      hipFuncAttributeMaxDynamicSharedMemorySize, (int)lds);
  attn_kernel<<<dim3(S_ / 16, H_, B_), 128, lds, stream>>>(Qb, Kb, Vt, mask, attn, ctxb);

  gemm_bf16_direct<<<dim3(D_ / 64, M_ / 128, 1), 128, 0, stream>>>(
      ctxb, Wto, bo, out, M_, H_ * DH_, D_, 3, 1.0f);
}
